// LinearAttention_40037685133426
// MI455X (gfx1250) — matrix operand order repaired and checked by reading, not compiled, not run on hardware
//
#include <hip/hip_runtime.h>
#include <hip/hip_bf16.h>

// Sizes (fixed by the reference)
#define B_    8
#define N_    4096
#define DIM_  512
#define HEADS_ 8
#define DHEAD_ 64
#define ROWS_ (B_ * N_)          // 32768
#define QKV_LD 1536              // 3*HIDDEN
#define SCALE_F 0.125f           // DHEAD^-0.5
#define EPS_ 1e-5f

typedef __bf16 bf16;
typedef __bf16 v16bf __attribute__((ext_vector_type(16)));
typedef float  v8f   __attribute__((ext_vector_type(8)));
typedef __bf16 v8bf  __attribute__((ext_vector_type(8)));   // 16-byte store chunk

__device__ __forceinline__ v8f wmma_bf16(v16bf a, v16bf b, v8f c) {
  // D = A(16x32 bf16) * B(32x16 bf16) + C(f32)
  return __builtin_amdgcn_wmma_f32_16x16x32_bf16(false, a, false, b, (short)0, c,
                                                 false, false);
}

// A fragment 16x32 from a row-major [M x K] tile (leading dim ld, in halves).
// Lane L(0..15): M=L, halves 0..15 -> K+0..15.
// Lane L(16..31): M=L-16, halves 0..15 -> K+16..31.
__device__ __forceinline__ v16bf load_a_rowmajor(const bf16* tile, int ld, int lane) {
  const bf16* p = tile + (size_t)(lane & 15) * ld + ((lane >> 4) << 4);
  v16bf a;
#pragma unroll
  for (int h = 0; h < 8; ++h) a[h] = p[h];
#pragma unroll
  for (int h = 0; h < 8; ++h) a[h + 8] = p[8 + h];
  return a;
}

// B fragment 32x16 where memory holds B^T row-major, i.e. W[n][k] and we need
// B[k][n] = W[n][k]. Lane L: N = L%16, halves h -> K = (L/16)*16 + h.
// => 16 contiguous bf16 per lane.
__device__ __forceinline__ v16bf load_b_from_wT(const bf16* wtile, int ld, int lane) {
  const bf16* p = wtile + (size_t)(lane & 15) * ld + ((lane >> 4) << 4);
  v16bf b;
#pragma unroll
  for (int h = 0; h < 16; ++h) b[h] = p[h];
  return b;
}

// ---------------------------------------------------------------------------
// K1: conditioning MLP: cond[b, j] = b_cond[j] + sum_i silu(c[b,i]) * w_cond[j,i]
// ---------------------------------------------------------------------------
__global__ void cond_kernel(const float* __restrict__ c,
                            const float* __restrict__ w_cond,
                            const float* __restrict__ b_cond,
                            float* __restrict__ cond) {
  int b = blockIdx.x;
  __shared__ float sc[512];
  for (int i = threadIdx.x; i < 512; i += blockDim.x) {
    float v = c[b * 512 + i];
    sc[i] = v / (1.0f + __expf(-v));
  }
  __syncthreads();
  for (int j = threadIdx.x; j < 1024; j += blockDim.x) {
    const float* w = w_cond + (size_t)j * 512;
    float acc = b_cond[j];
#pragma unroll 8
    for (int i = 0; i < 512; ++i) acc += sc[i] * w[i];
    cond[b * 1024 + j] = acc;
  }
}

// ---------------------------------------------------------------------------
// K2: convert weights fp32 -> bf16
// ---------------------------------------------------------------------------
__global__ void cvt_w_kernel(const float* __restrict__ wqkv,
                             const float* __restrict__ wout,
                             bf16* __restrict__ wq_b, bf16* __restrict__ wo_b) {
  int idx = blockIdx.x * blockDim.x + threadIdx.x;
  if (idx < 1536 * 512) wq_b[idx] = (bf16)wqkv[idx];
  if (idx < 512 * 512)  wo_b[idx] = (bf16)wout[idx];
}

// ---------------------------------------------------------------------------
// K3: fused LayerNorm + conditioning affine -> xn (bf16, [ROWS, 512])
// One 256-thread block per row.
// ---------------------------------------------------------------------------
__global__ void ln_affine_kernel(const float* __restrict__ x,
                                 const float* __restrict__ gamma_in,
                                 const float* __restrict__ cond,
                                 bf16* __restrict__ xn) {
  int row = blockIdx.x;
  int b = row >> 12;  // row / N_
  const float* xr = x + (size_t)row * DIM_;
  int t = threadIdx.x;
  float v0 = xr[t], v1 = xr[t + 256];
  float s = v0 + v1, ss = v0 * v0 + v1 * v1;
#pragma unroll
  for (int m = 16; m > 0; m >>= 1) {
    s += __shfl_xor(s, m, 32);
    ss += __shfl_xor(ss, m, 32);
  }
  __shared__ float red[2][8];
  int wid = t >> 5, lane = t & 31;
  if (lane == 0) { red[0][wid] = s; red[1][wid] = ss; }
  __syncthreads();
  if (t == 0) {
    float a = 0.f, q = 0.f;
#pragma unroll
    for (int i = 0; i < 8; ++i) { a += red[0][i]; q += red[1][i]; }
    red[0][0] = a; red[1][0] = q;
  }
  __syncthreads();
  float mu = red[0][0] * (1.0f / 512.0f);
  float var = red[1][0] * (1.0f / 512.0f) - mu * mu;
  float rstd = rsqrtf(var + EPS_);
  const float* scale = cond + b * 1024;
  const float* shift = scale + 512;
  bf16* o = xn + (size_t)row * DIM_;
  float y0 = (v0 - mu) * rstd * gamma_in[t]       * (scale[t] + 1.0f)       + shift[t];
  float y1 = (v1 - mu) * rstd * gamma_in[t + 256] * (scale[t + 256] + 1.0f) + shift[t + 256];
  o[t] = (bf16)y0;
  o[t + 256] = (bf16)y1;
}

// ---------------------------------------------------------------------------
// K4: QKV GEMM: qkv[m, j] = sum_k xn[m,k] * wq_b[j,k]   (bf16 out)
// Block: 256 thr (8 waves), tile 128(m) x 64(j); wave = 2x2 wmma tiles.
// Epilogue swizzles through LDS so global stores are contiguous b128 chunks.
// ---------------------------------------------------------------------------
__global__ void __launch_bounds__(256) gemm_qkv_kernel(const bf16* __restrict__ xn,
                                                       const bf16* __restrict__ wq,
                                                       bf16* __restrict__ qkv) {
  const int jblk = blockIdx.x * 64;
  const int mblk = blockIdx.y * 128;
  const int t = threadIdx.x, w = t >> 5, lane = t & 31;
  const int mt0 = (w >> 1) * 2;  // wave m-tile base (units of 16 rows)
  const int jt0 = (w & 1) * 2;   // wave j-tile base
  v8f acc[2][2] = {};
  for (int k = 0; k < 512; k += 32) {
    v16bf a0 = load_a_rowmajor(xn + (size_t)(mblk + mt0 * 16) * 512 + k, 512, lane);
    v16bf a1 = load_a_rowmajor(xn + (size_t)(mblk + (mt0 + 1) * 16) * 512 + k, 512, lane);
    v16bf b0 = load_b_from_wT(wq + (size_t)(jblk + jt0 * 16) * 512 + k, 512, lane);
    v16bf b1 = load_b_from_wT(wq + (size_t)(jblk + (jt0 + 1) * 16) * 512 + k, 512, lane);
    acc[0][0] = wmma_bf16(a0, b0, acc[0][0]);
    acc[0][1] = wmma_bf16(a0, b1, acc[0][1]);
    acc[1][0] = wmma_bf16(a1, b0, acc[1][0]);
    acc[1][1] = wmma_bf16(a1, b1, acc[1][1]);
  }
  // LDS swizzle: [128 rows][72-half stride] bf16
  __shared__ bf16 ob[128][72];
#pragma unroll
  for (int mi = 0; mi < 2; ++mi)
#pragma unroll
    for (int ji = 0; ji < 2; ++ji)
#pragma unroll
      for (int r = 0; r < 8; ++r) {
        int m = (mt0 + mi) * 16 + r + ((lane >> 4) << 3);
        int j = (jt0 + ji) * 16 + (lane & 15);
        ob[m][j] = (bf16)acc[mi][ji][r];
      }
  __syncthreads();
  // 128 rows x 8 chunks of 8 halves (16B) = 1024 chunks, 4 per thread
#pragma unroll
  for (int i = t; i < 1024; i += 256) {
    int row = i >> 3, cc = (i & 7) << 3;
    v8bf vchunk;
#pragma unroll
    for (int u = 0; u < 8; ++u) vchunk[u] = ob[row][cc + u];
    *(v8bf*)(qkv + (size_t)(mblk + row) * QKV_LD + jblk + cc) = vchunk;
  }
}

// ---------------------------------------------------------------------------
// K5: k softmax stats over the sequence dim: kmax[bh*64+d], kinvz = 1/sum exp
// One 256-thread block per (b,h); t%64 = d, t/64 = n-partition (4 parts).
// ---------------------------------------------------------------------------
__global__ void kstats_kernel(const bf16* __restrict__ qkv,
                              float* __restrict__ kmax, float* __restrict__ kinvz) {
  int bh = blockIdx.x;
  int b = bh >> 3, h = bh & 7;
  int t = threadIdx.x;
  int d = t & 63;
  int part = t >> 6;
  const bf16* base = qkv + (size_t)b * N_ * QKV_LD + 512 + h * 64 + d;
  __shared__ float red[4][64];
  float mx = -1e30f;
  for (int n = part * 1024; n < part * 1024 + 1024; ++n)
    mx = fmaxf(mx, (float)base[(size_t)n * QKV_LD]);
  red[part][d] = mx;
  __syncthreads();
  float m = fmaxf(fmaxf(red[0][d], red[1][d]), fmaxf(red[2][d], red[3][d]));
  __syncthreads();
  float s = 0.f;
  for (int n = part * 1024; n < part * 1024 + 1024; ++n)
    s += __expf((float)base[(size_t)n * QKV_LD] - m);
  red[part][d] = s;
  __syncthreads();
  if (part == 0) {
    float tot = red[0][d] + red[1][d] + red[2][d] + red[3][d];
    kmax[bh * 64 + d] = m;
    kinvz[bh * 64 + d] = 1.0f / tot;
  }
}

// ---------------------------------------------------------------------------
// K6: context[d,e] = (1/Z_d) * sum_n exp(k[n,d]-max_d) * v[n,e]  per (b,h)
// Block 512 thr (16 waves); each wave owns one 16x16 (d,e) tile; K-loop n+=32.
// ---------------------------------------------------------------------------
__global__ void __launch_bounds__(512) context_kernel(const bf16* __restrict__ qkv,
                                                      const float* __restrict__ kmax,
                                                      const float* __restrict__ kinvz,
                                                      bf16* __restrict__ ctx) {
  int bh = blockIdx.x;
  int b = bh >> 3, h = bh & 7;
  __shared__ bf16 kx[32][72];   // exp(k - max), padded (72 halves => conflict-free cols)
  __shared__ bf16 vv[32][72];
  __shared__ float smax[64], sinvz[64];
  int t = threadIdx.x, w = t >> 5, lane = t & 31;
  if (t < 64) { smax[t] = kmax[bh * 64 + t]; sinvz[t] = kinvz[bh * 64 + t]; }
  __syncthreads();
  const int dt = (w >> 2) * 16, et = (w & 3) * 16;
  const bf16* kbase = qkv + (size_t)b * N_ * QKV_LD + 512 + h * 64;
  const bf16* vbase = kbase + 512;
  v8f acc = {};
  int idx = t * 4, rr = idx >> 6, cc = idx & 63;  // 32 rows x 64 cols staging
  for (int n0 = 0; n0 < N_; n0 += 32) {
    const bf16* kp = kbase + (size_t)(n0 + rr) * QKV_LD + cc;
    const bf16* vp = vbase + (size_t)(n0 + rr) * QKV_LD + cc;
#pragma unroll
    for (int i = 0; i < 4; ++i) {
      kx[rr][cc + i] = (bf16)__expf((float)kp[i] - smax[cc + i]);
      vv[rr][cc + i] = vp[i];
    }
    __syncthreads();
    // A = kx^T (16x32): lane: m=d, halves map to local n
    v16bf a, bfrag;
    int d = dt + (lane & 15), nb = (lane >> 4) << 4;
#pragma unroll
    for (int hh = 0; hh < 16; ++hh) {
      int nl = hh + nb;
      a[hh] = kx[nl][d];
    }
    int e = et + (lane & 15), k0 = (lane >> 4) << 4;
#pragma unroll
    for (int hh = 0; hh < 16; ++hh) bfrag[hh] = vv[k0 + hh][e];
    acc = wmma_bf16(a, bfrag, acc);
    __syncthreads();
  }
#pragma unroll
  for (int r = 0; r < 8; ++r) {
    int d = dt + r + ((lane >> 4) << 3);
    int e = et + (lane & 15);
    ctx[((size_t)bh * 64 + d) * 64 + e] = (bf16)(acc[r] * sinvz[d]);
  }
}

// ---------------------------------------------------------------------------
// K7: out[n,e] = sum_d (softmax_d(q[n,:]) * scale_f)[d] * ctx[d,e]  per (b,h)
// Block 256 thr (8 waves) handles 128 rows; ao bf16 [ROWS, 512].
// Output staged in LDS, stored as contiguous b128 chunks.
// ---------------------------------------------------------------------------
__global__ void __launch_bounds__(256) attn_kernel(const bf16* __restrict__ qkv,
                                                   const bf16* __restrict__ ctx,
                                                   bf16* __restrict__ ao) {
  int bh = blockIdx.x;
  int b = bh >> 3, h = bh & 7;
  int nblk = blockIdx.y * 128;
  __shared__ bf16 ctxs[64][72];
  __shared__ bf16 qs[128][72];
  __shared__ bf16 ob[128][72];
  int t = threadIdx.x, w = t >> 5, lane = t & 31;
  for (int i = t; i < 4096; i += 256)
    ctxs[i >> 6][i & 63] = ctx[(size_t)bh * 4096 + i];
  if (t < 128) {
    const bf16* qr = qkv + ((size_t)b * N_ + nblk + t) * QKV_LD + h * 64;
    float vals[64];
    float mx = -1e30f;
#pragma unroll
    for (int i = 0; i < 64; ++i) { vals[i] = (float)qr[i]; mx = fmaxf(mx, vals[i]); }
    float s = 0.f;
#pragma unroll
    for (int i = 0; i < 64; ++i) { vals[i] = __expf(vals[i] - mx); s += vals[i]; }
    float inv = SCALE_F / s;
#pragma unroll
    for (int i = 0; i < 64; ++i) qs[t][i] = (bf16)(vals[i] * inv);
  }
  __syncthreads();
  v8f acc[4] = {};
  for (int kk = 0; kk < 64; kk += 32) {
    v16bf a = load_a_rowmajor(&qs[w * 16][0] + kk, 72, lane);
#pragma unroll
    for (int et = 0; et < 4; ++et) {
      v16bf bfrag;
      int e = et * 16 + (lane & 15), k0 = kk + ((lane >> 4) << 4);
#pragma unroll
      for (int hh = 0; hh < 16; ++hh) bfrag[hh] = ctxs[k0 + hh][e];
      acc[et] = wmma_bf16(a, bfrag, acc[et]);
    }
  }
#pragma unroll
  for (int et = 0; et < 4; ++et)
#pragma unroll
    for (int r = 0; r < 8; ++r) {
      int row = w * 16 + r + ((lane >> 4) << 3);
      int col = et * 16 + (lane & 15);
      ob[row][col] = (bf16)acc[et][r];
    }
  __syncthreads();
  // 128 rows x 8 chunks of 8 halves = 1024 chunks, 4 per thread
#pragma unroll
  for (int i = t; i < 1024; i += 256) {
    int row = i >> 3, cc = (i & 7) << 3;
    v8bf vchunk;
#pragma unroll
    for (int u = 0; u < 8; ++u) vchunk[u] = ob[row][cc + u];
    *(v8bf*)(ao + ((size_t)b * N_ + nblk + row) * 512 + h * 64 + cc) = vchunk;
  }
}

// ---------------------------------------------------------------------------
// K8: output projection + final LayerNorm, fused: block (512 thr, 16 waves)
// owns 16 full rows, so LN stats stay in LDS. out fp32 to d_out.
// ---------------------------------------------------------------------------
__global__ void __launch_bounds__(512) out_ln_kernel(const bf16* __restrict__ ao,
                                                     const bf16* __restrict__ wo,
                                                     const float* __restrict__ gamma_out,
                                                     float* __restrict__ out) {
  int m0 = blockIdx.x * 16;
  int t = threadIdx.x, w = t >> 5, lane = t & 31;
  __shared__ float ybuf[16][516];  // pad 4 floats
  int jt = w * 2;  // wave covers cols jt*16 .. jt*16+31
  v8f acc[2] = {};
  for (int k = 0; k < 512; k += 32) {
    v16bf a = load_a_rowmajor(ao + (size_t)m0 * 512 + k, 512, lane);
#pragma unroll
    for (int ji = 0; ji < 2; ++ji) {
      v16bf bfrag = load_b_from_wT(wo + (size_t)((jt + ji) * 16) * 512 + k, 512, lane);
      acc[ji] = wmma_bf16(a, bfrag, acc[ji]);
    }
  }
#pragma unroll
  for (int ji = 0; ji < 2; ++ji)
#pragma unroll
    for (int r = 0; r < 8; ++r) {
      int m = r + ((lane >> 4) << 3);
      int j = (jt + ji) * 16 + (lane & 15);
      ybuf[m][j] = acc[ji][r];
    }
  __syncthreads();
  // wave w normalizes row w: each lane owns 16 cols
  int row = w;
  float vals[16], s = 0.f, ss = 0.f;
#pragma unroll
  for (int i = 0; i < 16; ++i) {
    float v = ybuf[row][lane * 16 + i];
    vals[i] = v; s += v; ss += v * v;
  }
#pragma unroll
  for (int m = 16; m > 0; m >>= 1) {
    s += __shfl_xor(s, m, 32);
    ss += __shfl_xor(ss, m, 32);
  }
  float mu = s * (1.0f / 512.0f);
  float var = ss * (1.0f / 512.0f) - mu * mu;
  float rstd = rsqrtf(var + EPS_);
#pragma unroll
  for (int i = 0; i < 16; ++i) {
    int c = lane * 16 + i;
    out[(size_t)(m0 + row) * 512 + c] = (vals[i] - mu) * rstd * gamma_out[c];
  }
}

// ---------------------------------------------------------------------------
extern "C" void kernel_launch(void* const* d_in, const int* in_sizes, int n_in,
                              void* d_out, int out_size, void* d_ws, size_t ws_size,
                              hipStream_t stream) {
  const float* x    = (const float*)d_in[0];
  const float* ce   = (const float*)d_in[1];
  const float* gin  = (const float*)d_in[2];
  const float* wc   = (const float*)d_in[3];
  const float* bc   = (const float*)d_in[4];
  const float* wqkv = (const float*)d_in[5];
  const float* wout = (const float*)d_in[6];
  const float* gout = (const float*)d_in[7];
  float* out = (float*)d_out;

  char* p = (char*)d_ws;
  float* cond  = (float*)p;  p += (size_t)B_ * 1024 * 4;                 // 32 KB
  float* kmax  = (float*)p;  p += (size_t)B_ * HEADS_ * 64 * 4;          // 16 KB
  float* kinvz = (float*)p;  p += (size_t)B_ * HEADS_ * 64 * 4;          // 16 KB
  bf16* wq_b   = (bf16*)p;   p += (size_t)1536 * 512 * 2;                // 1.5 MB
  bf16* wo_b   = (bf16*)p;   p += (size_t)512 * 512 * 2;                 // 0.5 MB
  bf16* ctx    = (bf16*)p;   p += (size_t)B_ * HEADS_ * 64 * 64 * 2;     // 0.5 MB
  bf16* xn     = (bf16*)p;   p += (size_t)ROWS_ * 512 * 2;               // 32 MB
  bf16* qkv    = (bf16*)p;   p += (size_t)ROWS_ * QKV_LD * 2;            // 96 MB
  bf16* ao     = (bf16*)p;   p += (size_t)ROWS_ * 512 * 2;               // 32 MB

  cond_kernel<<<B_, 256, 0, stream>>>(ce, wc, bc, cond);
  cvt_w_kernel<<<(1536 * 512 + 255) / 256, 256, 0, stream>>>(wqkv, wout, wq_b, wo_b);
  ln_affine_kernel<<<ROWS_, 256, 0, stream>>>(x, gin, cond, xn);
  gemm_qkv_kernel<<<dim3(1536 / 64, ROWS_ / 128), 256, 0, stream>>>(xn, wq_b, qkv);
  kstats_kernel<<<B_ * HEADS_, 256, 0, stream>>>(qkv, kmax, kinvz);
  context_kernel<<<B_ * HEADS_, 512, 0, stream>>>(qkv, kmax, kinvz, ctx);
  attn_kernel<<<dim3(B_ * HEADS_, N_ / 128), 256, 0, stream>>>(qkv, ctx, ao);
  out_ln_kernel<<<ROWS_ / 16, 512, 0, stream>>>(ao, wo_b, gout, out);
}
